// NeuralNetworkModel_tanh_aniso_wm_76038101008853
// MI455X (gfx1250) — compile-verified, matrix operand order checked by reading, not run on hardware
//
#include <hip/hip_runtime.h>
#include <math.h>

// Problem constants (from reference)
#define NROWS 262144
#define HDIM  256
#define NIX   9
#define NIY   6
#define HHALF 128

#define MTILE 64     // rows per workgroup
#define AS    264    // LDS stride (bf16 elems) for H1 hi/lo (16B-aligned, bank-padded)
#define HS    260    // LDS stride (f32 elems) for H3/pp-qq buffer
#define BS    40     // staged-B column stride (bf16 elems) = 80B -> 2-way banks at worst

typedef __attribute__((ext_vector_type(16))) __bf16 v16bf;
typedef __attribute__((ext_vector_type(8)))  __bf16 v8bf;
typedef __attribute__((ext_vector_type(8)))  float  v8f;
typedef __attribute__((ext_vector_type(4)))  unsigned int v4u;

static __device__ __forceinline__ unsigned short f2bf_rne(float f) {
    unsigned int u = __float_as_uint(f);
    u += 0x7FFFu + ((u >> 16) & 1u);
    return (unsigned short)(u >> 16);
}
static __device__ __forceinline__ float bf2f(unsigned short h) {
    return __uint_as_float(((unsigned int)h) << 16);
}
static __device__ __forceinline__ float elu2(float pos, float g) {
    // relu(pos) - relu(-(exp(g)-1)) == max(pos,0) + min(exp(g)-1, 0)
    return fmaxf(pos, 0.f) + fminf(expf(g) - 1.f, 0.f);
}

// ---------------------------------------------------------------------------
// Prep: W3 (256x256 f32, row-major k x n) -> transposed split-bf16 planes
// w3t_hi/lo[n*256 + k]; B-fragment lanes then read contiguous 16B runs of K.
// ---------------------------------------------------------------------------
__global__ void prep_w3_kernel(const float* __restrict__ W3,
                               unsigned short* __restrict__ w3t_hi,
                               unsigned short* __restrict__ w3t_lo) {
    int idx = blockIdx.x * blockDim.x + threadIdx.x;   // 0..65535
    int k = idx >> 8;
    int n = idx & 255;
    float v = W3[idx];
    unsigned short hi = f2bf_rne(v);
    float lo = v - bf2f(hi);
    w3t_hi[n * 256 + k] = hi;
    w3t_lo[n * 256 + k] = f2bf_rne(lo);
}

// ---------------------------------------------------------------------------
// Fused kernel. 256 threads = 8 waves; wave w: rows 16*(w>>1), cols 128*(w&1).
// B K-chunks (32xK, all 256 cols, hi+lo = 32KB) are staged into LDS once per
// workgroup per chunk, double-buffered, so global B traffic is 256KB/WG total.
// ---------------------------------------------------------------------------
__global__ __launch_bounds__(256) void fused_mlp_kernel(
    const float* __restrict__ x,  const float* __restrict__ y,
    const float* __restrict__ W1, const float* __restrict__ W2,
    const float* __restrict__ Wp, const float* __restrict__ Wq,
    const unsigned short* __restrict__ w3t_hi,
    const unsigned short* __restrict__ w3t_lo,
    float* __restrict__ out)
{
    __shared__ __align__(16) unsigned short s_h1hi[MTILE * AS];     //  33792 B
    __shared__ __align__(16) unsigned short s_h1lo[MTILE * AS];     //  33792 B
    __shared__ __align__(16) float          s_h3[MTILE * HS];       //  66560 B
    __shared__ __align__(16) unsigned short s_bst[2 * 2 * HDIM * BS]; // 81920 B
    __shared__ __align__(16) float          s_red[MTILE * 4 * 9];   //   9216 B

    const int t     = threadIdx.x;
    const int rbase = blockIdx.x * MTILE;

    // Cooperative stage of B K-chunk kc (hi+lo planes, 256 cols x 32 K) into buf.
    auto stageB = [&](int buf, int kc) {
        #pragma unroll
        for (int it = 0; it < 8; ++it) {
            int fi    = it * 256 + t;        // 0..2047 sixteen-byte chunks
            int plane = fi >> 10;            // 0=hi, 1=lo (1024 chunks per plane)
            int rem   = fi & 1023;
            int col   = rem >> 2;            // 0..255
            int ch    = rem & 3;             // 4 x 16B per column
            const unsigned short* src =
                (plane ? w3t_lo : w3t_hi) + col * HDIM + kc * 32 + ch * 8;
            v4u v = *(const v4u*)src;
            *(v4u*)&s_bst[((buf * 2 + plane) * HDIM + col) * BS + ch * 8] = v;
        }
    };

    // ---- Stage 1: H1 (split bf16) and H3 (f32) tiles into LDS ----
    {
        float w1c[NIX], w2c[NIY];
        #pragma unroll
        for (int i = 0; i < NIX; ++i) w1c[i] = W1[i * HDIM + t];
        #pragma unroll
        for (int i = 0; i < NIY; ++i) w2c[i] = W2[i * HDIM + t];

        for (int r = 0; r < MTILE; ++r) {
            const float* xr = x + (size_t)(rbase + r) * NIX;   // uniform -> scalar loads
            float h = 0.f;
            #pragma unroll
            for (int i = 0; i < NIX; ++i) h = fmaf(xr[i], w1c[i], h);
            h = tanhf(h);
            unsigned short hh = f2bf_rne(h);
            s_h1hi[r * AS + t] = hh;
            s_h1lo[r * AS + t] = f2bf_rne(h - bf2f(hh));

            const float* yr = y + (size_t)(rbase + r) * NIY;
            float g = 0.f;
            #pragma unroll
            for (int i = 0; i < NIY; ++i) g = fmaf(yr[i], w2c[i], g);
            s_h3[r * HS + t] = tanhf(g);
        }
    }
    stageB(0, 0);          // prefetch first B chunk alongside stage-1 publication
    __syncthreads();

    // ---- Stage 2: 16x128 tile per wave, split-bf16 WMMA, LDS-staged B ----
    {
        const int wave    = t >> 5;
        const int lane    = t & 31;
        const int l16     = lane & 15;
        const int hi16    = lane >> 4;        // K-offset selector per A/B layout
        const int rowgrp  = wave >> 1;        // 0..3
        const int colhalf = wave & 1;         // 0..1
        const int m       = rowgrp * 16 + l16;
        const int koff    = hi16 * 8;

        v8f acc[8] = {};

        for (int kc = 0; kc < 8; ++kc) {
            const int buf = kc & 1;
            const int k0  = kc * 32 + koff;

            union { v16bf v; v8bf h[2]; } ua_hi, ua_lo;
            ua_hi.h[0] = *(const v8bf*)&s_h1hi[m * AS + k0];
            ua_hi.h[1] = *(const v8bf*)&s_h1hi[m * AS + k0 + 16];
            ua_lo.h[0] = *(const v8bf*)&s_h1lo[m * AS + k0];
            ua_lo.h[1] = *(const v8bf*)&s_h1lo[m * AS + k0 + 16];

            if (kc < 7) stageB(buf ^ 1, kc + 1);   // prefetch next chunk

            #pragma unroll
            for (int nt = 0; nt < 8; ++nt) {
                const int ncol = colhalf * 128 + nt * 16 + l16;
                const int bh = ((buf * 2 + 0) * HDIM + ncol) * BS + koff;
                const int bl = ((buf * 2 + 1) * HDIM + ncol) * BS + koff;
                union { v16bf v; v8bf h[2]; } ub_hi, ub_lo;
                ub_hi.h[0] = *(const v8bf*)&s_bst[bh];
                ub_hi.h[1] = *(const v8bf*)&s_bst[bh + 16];
                ub_lo.h[0] = *(const v8bf*)&s_bst[bl];
                ub_lo.h[1] = *(const v8bf*)&s_bst[bl + 16];

                acc[nt] = __builtin_amdgcn_wmma_f32_16x16x32_bf16(
                    false, ua_hi.v, false, ub_hi.v, (short)0, acc[nt], false, false);
                acc[nt] = __builtin_amdgcn_wmma_f32_16x16x32_bf16(
                    false, ua_lo.v, false, ub_hi.v, (short)0, acc[nt], false, false);
                acc[nt] = __builtin_amdgcn_wmma_f32_16x16x32_bf16(
                    false, ua_hi.v, false, ub_lo.v, (short)0, acc[nt], false, false);
            }
            __syncthreads();   // publish next buffer; fence reads of current one
        }

        // H2 = tanh(acc); fuse pp/qq = H2 * H3 in place (unique owner per element)
        #pragma unroll
        for (int nt = 0; nt < 8; ++nt) {
            const int col = colhalf * 128 + nt * 16 + l16;
            #pragma unroll
            for (int g = 0; g < 8; ++g) {
                const int row = rowgrp * 16 + g + hi16 * 8;   // C layout: lanes>=16 -> M=8+g
                s_h3[row * HS + col] *= tanhf(acc[nt][g]);
            }
        }
    }
    __syncthreads();

    // ---- Stage 3: op = pp@Wp (5 cols), oq = qq@Wq (4 cols); 4-way split ----
    {
        const int row = t >> 2;
        const int q   = t & 3;
        float ap[5] = {0.f, 0.f, 0.f, 0.f, 0.f};
        float aq[4] = {0.f, 0.f, 0.f, 0.f};
        for (int c = q * 32; c < q * 32 + 32; ++c) {
            const float ppv = s_h3[row * HS + c];
            const float qqv = s_h3[row * HS + HHALF + c];
            #pragma unroll
            for (int j = 0; j < 5; ++j) ap[j] = fmaf(ppv, Wp[c * 5 + j], ap[j]);
            #pragma unroll
            for (int j = 0; j < 4; ++j) aq[j] = fmaf(qqv, Wq[c * 4 + j], aq[j]);
        }
        float* r = &s_red[(row * 4 + q) * 9];
        #pragma unroll
        for (int j = 0; j < 5; ++j) r[j] = ap[j];
        #pragma unroll
        for (int j = 0; j < 4; ++j) r[5 + j] = aq[j];
    }
    __syncthreads();

    // ---- Stage 4: combine partials, ELU epilogue, store 9 outputs/row ----
    if (t < MTILE) {
        float v[9];
        #pragma unroll
        for (int j = 0; j < 9; ++j)
            v[j] = s_red[(t * 4 + 0) * 9 + j] + s_red[(t * 4 + 1) * 9 + j]
                 + s_red[(t * 4 + 2) * 9 + j] + s_red[(t * 4 + 3) * 9 + j];
        const float f1 = v[0], f2 = v[1], f3 = v[2], wm1 = v[3], wm2 = v[4];
        const float f4 = v[5], f5 = v[6], wm3 = v[7], wm4 = v[8];
        float* o = out + (size_t)(rbase + t) * 9;
        o[0] = elu2(f1, f1);
        o[1] = elu2(f2, f2);
        o[2] = elu2(f2, f3);
        o[3] = elu2(f2, f4);
        o[4] = elu2(f2, f5);
        o[5] = wm1; o[6] = wm2; o[7] = wm3; o[8] = wm4;
    }
}

extern "C" void kernel_launch(void* const* d_in, const int* in_sizes, int n_in,
                              void* d_out, int out_size, void* d_ws, size_t ws_size,
                              hipStream_t stream) {
    (void)in_sizes; (void)n_in; (void)out_size; (void)ws_size;
    const float* x  = (const float*)d_in[0];
    const float* y  = (const float*)d_in[1];
    const float* W1 = (const float*)d_in[2];
    const float* W2 = (const float*)d_in[3];
    const float* W3 = (const float*)d_in[4];
    const float* Wp = (const float*)d_in[5];
    const float* Wq = (const float*)d_in[6];

    unsigned short* w3t_hi = (unsigned short*)d_ws;          // 128 KB
    unsigned short* w3t_lo = w3t_hi + HDIM * HDIM;           // 128 KB

    prep_w3_kernel<<<(HDIM * HDIM) / 256, 256, 0, stream>>>(W3, w3t_hi, w3t_lo);
    fused_mlp_kernel<<<NROWS / MTILE, 256, 0, stream>>>(
        x, y, W1, W2, Wp, Wq, w3t_hi, w3t_lo, (float*)d_out);
}
